// CausalBankModel_35974646071625
// MI455X (gfx1250) — matrix operand order repaired and checked by reading, not compiled, not run on hardware
//
#include <hip/hip_runtime.h>
#include <math.h>

// ---------------- model dims ----------------
#define V_   32000
#define D_   256
#define M_   256
#define W_   8
#define HL_  1024
#define S_   1024
#define B_   2
#define ROWS (B_ * S_)          // 2048
#define SPAD (S_ + W_ - 1)      // 1031 padded rows per batch for window branch

typedef __attribute__((ext_vector_type(16))) _Float16 v16h;
typedef __attribute__((ext_vector_type(8)))  _Float16 v8h;
typedef __attribute__((ext_vector_type(8)))  float    v8f;

union Frag { v16h v; v8h h[2]; };

// ---------------------------------------------------------------------------
// WMMA GEMM: C[M x N] = act(A[M x K](f16) * B^T[N x K](f16) + bias)
// block = 256 threads (8 wave32s, 2x4), block tile 64 x 256.
// wave tile 32 x 64 = 2x4 subtiles of 16x16 -> 8 accumulators;
// per K-step: 12 b128 loads feed 8 WMMAs (A reused 4x, B reused 2x).
// A row address = A + m*rsA + (m>>10)*beA   (beA handles the padded window buf)
// ---------------------------------------------------------------------------
__global__ __launch_bounds__(256)
void wmma_gemm(const _Float16* __restrict__ A, int rsA, int beA,
               const _Float16* __restrict__ BT, int K, int N,
               const float* __restrict__ bias, int relu,
               float* __restrict__ outF32, _Float16* __restrict__ outF16)
{
    const int lane = threadIdx.x & 31;
    const int wave = threadIdx.x >> 5;
    const int wm   = wave >> 2, wn = wave & 3;
    const int hi   = lane >> 4, nl = lane & 15;
    const int tileM = blockIdx.y * 64  + wm * 32;
    const int tileN = blockIdx.x * 256 + wn * 64;

    const _Float16* Arow[2];
    #pragma unroll
    for (int i = 0; i < 2; ++i) {
        const int m = tileM + i * 16 + nl;       // A fragment row for this lane
        Arow[i] = A + (size_t)m * rsA + (size_t)(m >> 10) * beA;
    }
    const _Float16* Brow[4];
    #pragma unroll
    for (int j = 0; j < 4; ++j)                  // B^T: contiguous K per lane
        Brow[j] = BT + (size_t)(tileN + j * 16 + nl) * K;

    v8f acc[2][4];
    #pragma unroll
    for (int i = 0; i < 2; ++i)
        #pragma unroll
        for (int j = 0; j < 4; ++j) { v8f z = {}; acc[i][j] = z; }

    for (int kb = 0; kb < K; kb += 32) {
        Frag a[2], b[4];
        #pragma unroll
        for (int i = 0; i < 2; ++i) {
            // A 16x32 f16 layout: K runs [hi*8, hi*8+8) and [16+hi*8, ...)
            a[i].h[0] = *(const v8h*)(Arow[i] + kb + hi * 8);
            a[i].h[1] = *(const v8h*)(Arow[i] + kb + 16 + hi * 8);
        }
        #pragma unroll
        for (int j = 0; j < 4; ++j) {
            // B 32x16 f16 layout: K runs [hi*8, hi*8+8) and [16+hi*8, ...)
            b[j].h[0] = *(const v8h*)(Brow[j] + kb + hi * 8);
            b[j].h[1] = *(const v8h*)(Brow[j] + kb + 16 + hi * 8);
        }
        __builtin_prefetch(Brow[0] + kb + 256, 0, 1);   // global_prefetch_b8
        __builtin_prefetch(Brow[2] + kb + 256, 0, 1);
        #pragma unroll
        for (int i = 0; i < 2; ++i)
            #pragma unroll
            for (int j = 0; j < 4; ++j)
                acc[i][j] = __builtin_amdgcn_wmma_f32_16x16x32_f16(
                    false, a[i].v, false, b[j].v, (short)0, acc[i][j], false, false);
    }

    #pragma unroll
    for (int j = 0; j < 4; ++j) {
        const int n  = tileN + j * 16 + nl;
        const float bn = bias ? bias[n] : 0.0f;
        #pragma unroll
        for (int i = 0; i < 2; ++i) {
            #pragma unroll
            for (int r = 0; r < 8; ++r) {
                // C/D layout: lane col = nl, rows m = hi*8 + r
                const int mr = tileM + i * 16 + hi * 8 + r;
                float v = acc[i][j][r] + bn;
                if (relu) v = v > 0.0f ? v : 0.0f;
                const size_t o = (size_t)mr * N + n;
                if (outF32) outF32[o] = v;
                if (outF16) outF16[o] = (_Float16)v;
            }
        }
    }
}

// -------- zero the 7 leading pad rows of each batch in the window buffer ----
__global__ void padzero_kernel(_Float16* __restrict__ xhp)
{
    int i = blockIdx.x * blockDim.x + threadIdx.x;
    const int per = (W_ - 1) * D_;               // 1792 pad elements per batch
    if (i < B_ * per) {
        int b = i / per, j = i % per;
        xhp[(size_t)b * SPAD * D_ + j] = (_Float16)0.0f;
    }
}

// -------- embedding gather; emit f16 x into 3 consumers ---------------------
__global__ __launch_bounds__(256)
void gather_kernel(const int* __restrict__ tokens, const float* __restrict__ emb,
                   _Float16* __restrict__ xh, _Float16* __restrict__ xhp,
                   _Float16* __restrict__ feat)
{
    const int r = blockIdx.x;          // 0..2047 = b*1024 + s
    const int d = threadIdx.x;         // 0..255
    const int b = r >> 10, s = r & (S_ - 1);
    const int tok = tokens[r];
    const float v = emb[(size_t)tok * D_ + d];
    const _Float16 h = (_Float16)v;
    xh  [(size_t)r * D_ + d]                           = h;  // A of GEMM1
    xhp [((size_t)(b * SPAD + s + (W_ - 1))) * D_ + d] = h;  // window branch A
    feat[(size_t)r * (M_ + D_) + M_ + d]               = h;  // feat cols 256..511
}

// -------- tiled f32 -> f16 convert + transpose (K x N -> N x K) -------------
__global__ __launch_bounds__(256)
void t16_kernel(const float* __restrict__ src, _Float16* __restrict__ dst,
                int K, int N)
{
    __shared__ _Float16 tile[32][33];
    const int kb = blockIdx.y * 32, nb = blockIdx.x * 32;
    const int tx = threadIdx.x & 31, ty = threadIdx.x >> 5;  // 8 rows per pass
    #pragma unroll
    for (int yy = 0; yy < 32; yy += 8)
        tile[ty + yy][tx] = (_Float16)src[(size_t)(kb + ty + yy) * N + nb + tx];
    __syncthreads();
    #pragma unroll
    for (int yy = 0; yy < 32; yy += 8)
        dst[(size_t)(nb + ty + yy) * K + kb + tx] = tile[tx][ty + yy];
}

// -------- sequential decay scan: states[t] = a*states[t-1] + u[t] -----------
__global__ void scan_kernel(const float* __restrict__ u,
                            const float* __restrict__ decays,
                            _Float16* __restrict__ feat)
{
    const int t = blockIdx.x * blockDim.x + threadIdx.x;  // 0..511 = b*256 + m
    if (t >= B_ * M_) return;
    const int b = t >> 8, m = t & (M_ - 1);
    const float a = decays[m];
    float s = 0.0f;
    const float* up = u + (size_t)(b << 10) * M_ + m;
    _Float16*    fp = feat + (size_t)(b << 10) * (M_ + D_) + m;
    for (int i = 0; i < S_; ++i) {
        s = fmaf(a, s, up[(size_t)i * M_]);
        fp[(size_t)i * (M_ + D_)] = (_Float16)s;          // feat cols 0..255
    }
}

// -------- deterministic per-row stats: mean / max / std over V --------------
__global__ __launch_bounds__(256)
void stats_kernel(const float* __restrict__ lin, const float* __restrict__ loc,
                  float* __restrict__ stats)
{
    const int id     = blockIdx.x;          // 0..4095
    const int branch = id >> 11;
    const int row    = id & (ROWS - 1);
    const float4* p4 = (const float4*)(((branch ? loc : lin) + (size_t)row * V_));
    float sum = 0.0f, sq = 0.0f, mx = -3.4e38f;
    for (int i = threadIdx.x; i < V_ / 4; i += 256) {
        const float4 v = p4[i];
        sum += (v.x + v.y) + (v.z + v.w);
        sq  += (v.x * v.x + v.y * v.y) + (v.z * v.z + v.w * v.w);
        mx   = fmaxf(mx, fmaxf(fmaxf(v.x, v.y), fmaxf(v.z, v.w)));
    }
    __shared__ float ssum[256], ssq[256], smx[256];
    ssum[threadIdx.x] = sum; ssq[threadIdx.x] = sq; smx[threadIdx.x] = mx;
    __syncthreads();
    for (int st = 128; st > 0; st >>= 1) {
        if (threadIdx.x < st) {
            ssum[threadIdx.x] += ssum[threadIdx.x + st];
            ssq [threadIdx.x] += ssq [threadIdx.x + st];
            smx [threadIdx.x]  = fmaxf(smx[threadIdx.x], smx[threadIdx.x + st]);
        }
        __syncthreads();
    }
    if (threadIdx.x == 0) {
        const float mean = ssum[0] / (float)V_;
        const float var  = ssq[0] / (float)V_ - mean * mean;
        float* o = stats + (size_t)(branch * ROWS + row) * 4;
        o[0] = mean; o[1] = smx[0]; o[2] = sqrtf(fmaxf(var, 0.0f));
    }
}

// -------- gate + blend (lin logits live in d_out, blended in place) ---------
__global__ __launch_bounds__(256)
void blend_kernel(float* __restrict__ out, const float* __restrict__ loc,
                  const float* __restrict__ stats,
                  const float* __restrict__ gw, const float* __restrict__ gb)
{
    const int row = blockIdx.x;
    const float* s0 = stats + (size_t)row * 4;            // lin stats
    const float* s1 = stats + (size_t)(ROWS + row) * 4;   // loc stats
    const float z = gw[0]*s0[0] + gw[1]*s0[1] + gw[2]*s0[2]
                  + gw[3]*s1[0] + gw[4]*s1[1] + gw[5]*s1[2] + gb[0];
    const float g = 1.0f / (1.0f + expf(-z));
    float4*       po4 = (float4*)(out + (size_t)row * V_);
    const float4* pl4 = (const float4*)(loc + (size_t)row * V_);
    for (int i = threadIdx.x; i < V_ / 4; i += 256) {
        float4 a = po4[i];
        const float4 b = pl4[i];
        a.x = g * a.x + (1.0f - g) * b.x;
        a.y = g * a.y + (1.0f - g) * b.y;
        a.z = g * a.z + (1.0f - g) * b.z;
        a.w = g * a.w + (1.0f - g) * b.w;
        po4[i] = a;
    }
}

// ---------------------------------------------------------------------------
extern "C" void kernel_launch(void* const* d_in, const int* in_sizes, int n_in,
                              void* d_out, int out_size, void* d_ws, size_t ws_size,
                              hipStream_t stream)
{
    (void)in_sizes; (void)n_in; (void)out_size; (void)ws_size;
    const int*   tokens  = (const int*)  d_in[0];
    const float* emb     = (const float*)d_in[1];
    const float* in_proj = (const float*)d_in[2];
    const float* decays  = (const float*)d_in[3];
    const float* w1      = (const float*)d_in[4];
    const float* b1      = (const float*)d_in[5];
    const float* w2      = (const float*)d_in[6];
    const float* b2      = (const float*)d_in[7];
    const float* lw1     = (const float*)d_in[8];
    const float* lb1     = (const float*)d_in[9];
    const float* lw2     = (const float*)d_in[10];
    const float* lb2     = (const float*)d_in[11];
    const float* gw      = (const float*)d_in[12];
    const float* gb      = (const float*)d_in[13];
    float* out = (float*)d_out;

    char* base = (char*)d_ws;
    size_t off = 0;
    auto take = [&](size_t bytes) -> char* {
        off = (off + 255) & ~(size_t)255;
        char* p = base + off; off += bytes; return p;
    };
    float*    locL = (float*)   take((size_t)ROWS * V_ * 4);        // loc logits
    _Float16* w2T  = (_Float16*)take((size_t)V_ * HL_ * 2);
    _Float16* lw2T = (_Float16*)take((size_t)V_ * HL_ * 2);
    _Float16* w1T  = (_Float16*)take((size_t)HL_ * (M_ + D_) * 2);
    _Float16* lw1T = (_Float16*)take((size_t)HL_ * (W_ * D_) * 2);
    _Float16* ipT  = (_Float16*)take((size_t)M_ * D_ * 2);
    _Float16* xh   = (_Float16*)take((size_t)ROWS * D_ * 2);
    _Float16* xhp  = (_Float16*)take((size_t)B_ * SPAD * D_ * 2);
    _Float16* feat = (_Float16*)take((size_t)ROWS * (M_ + D_) * 2);
    _Float16* h1   = (_Float16*)take((size_t)ROWS * HL_ * 2);
    _Float16* h2   = (_Float16*)take((size_t)ROWS * HL_ * 2);
    float*    u    = (float*)   take((size_t)ROWS * M_ * 4);
    float*    st   = (float*)   take((size_t)2 * ROWS * 4 * 4);

    // stage 0: gather + weight convert/transpose
    padzero_kernel<<<14, 256, 0, stream>>>(xhp);
    gather_kernel<<<ROWS, 256, 0, stream>>>(tokens, emb, xh, xhp, feat);
    t16_kernel<<<dim3(M_/32,        D_/32),        256, 0, stream>>>(in_proj, ipT,  D_,      M_);
    t16_kernel<<<dim3(HL_/32,      (M_+D_)/32),    256, 0, stream>>>(w1,      w1T,  M_+D_,   HL_);
    t16_kernel<<<dim3(HL_/32,      (W_*D_)/32),    256, 0, stream>>>(lw1,     lw1T, W_*D_,   HL_);
    t16_kernel<<<dim3(V_/32,        HL_/32),       256, 0, stream>>>(w2,      w2T,  HL_,     V_);
    t16_kernel<<<dim3(V_/32,        HL_/32),       256, 0, stream>>>(lw2,     lw2T, HL_,     V_);

    // stage 1: u = x @ in_proj  (f32 out for scan precision)
    wmma_gemm<<<dim3(M_/256, ROWS/64), 256, 0, stream>>>(
        xh, D_, 0, ipT, D_, M_, nullptr, 0, u, nullptr);
    // stage 2: decay-bank scan -> feat cols [0,256)
    scan_kernel<<<2, 256, 0, stream>>>(u, decays, feat);
    // stage 3: h = relu(feat @ w1 + b1)
    wmma_gemm<<<dim3(HL_/256, ROWS/64), 256, 0, stream>>>(
        feat, M_ + D_, 0, w1T, M_ + D_, HL_, b1, 1, nullptr, h1);
    // stage 4: h2 = relu(loc @ lw1 + lb1); overlapping windows via rsA=256, beA=7*256
    wmma_gemm<<<dim3(HL_/256, ROWS/64), 256, 0, stream>>>(
        xhp, D_, (SPAD - S_) * D_, lw1T, W_ * D_, HL_, lb1, 1, nullptr, h2);
    // stage 5: the two big logits GEMMs (f16 weights resident in 192MB L2)
    wmma_gemm<<<dim3(V_/256, ROWS/64), 256, 0, stream>>>(
        h1, HL_, 0, w2T, HL_, V_, b2, 0, out, nullptr);
    wmma_gemm<<<dim3(V_/256, ROWS/64), 256, 0, stream>>>(
        h2, HL_, 0, lw2T, HL_, V_, lb2, 0, locL, nullptr);
    // stage 6: per-row stats + gated blend
    stats_kernel<<<2 * ROWS, 256, 0, stream>>>(out, locL, st);
    blend_kernel<<<ROWS, 256, 0, stream>>>(out, locL, st, gw, gb);
}
